// Xception_60249801229099
// MI455X (gfx1250) — matrix operand order repaired and checked by reading, not compiled, not run on hardware
//
#include <hip/hip_runtime.h>

// ---------------------------------------------------------------------------
// Xception forward for MI455X (gfx1250, wave32).
// Heavy compute: bf16 WMMA implicit-GEMM (v_wmma_f32_16x16x32_bf16, fp32
// accum) with BN/bias/residual/ReLU fused in the epilogue.
//  - k_gemm_pw : 1x1 convs + stride-2 shortcuts. Block = 8 waves sharing one
//                64-col n-block; B tile cooperatively staged in LDS (double
//                buffered, ds_load_b128 fragment reads) so weights hit L2
//                once per block instead of once per wave. Each wave: 32x64
//                tile; all 4 B fragments preloaded -> one dscnt wait and
//                8 back-to-back WMMAs per k-step; prefetch on the A stream.
//  - k_conv3x3 : conv2 (3x3, pad 1) as 9-tap accumulated GEMM (~5% FLOPs).
// Activations: bf16 NHWC, channel stride padded to multiple of 32
// (728 -> 736) with hard zeros. Workspace: 48MB bf16 weight arena,
// 3x136MB + 40MB activation slabs, small fp32 pooled buffer (~497MB).
// ---------------------------------------------------------------------------

typedef __attribute__((ext_vector_type(16))) __bf16        v16bf;
typedef __attribute__((ext_vector_type(8)))  float         v8f;
typedef __attribute__((ext_vector_type(4)))  unsigned int  u32x4;
typedef unsigned short u16;

__device__ __forceinline__ float bf2f(u16 h) {
    return __uint_as_float(((unsigned int)h) << 16);
}
__device__ __forceinline__ u16 f2bf(float f) {
    unsigned int u = __float_as_uint(f);
    u += 0x7fffu + ((u >> 16) & 1u);   // round-to-nearest-even
    return (u16)(u >> 16);
}

union FragBF { v16bf v; u32x4 q[2]; };

// LDS B-tile row stride in u16 elements: 32 k-elements + pad to 40 (80B) to
// spread bank groups while keeping every fragment/stage access 16B-aligned.
#define BSTR 40

// ---------------------------------------------------------------------------
// 1x1 conv GEMM (pointwise / strided shortcut), LDS-staged weights.
// A: bf16 NHWC [NB,Hin,Win,Cinp]; W: bf16 [Coutp][Cinp]; O/Res: NHWC [M,Coutp].
// Fragment layouts per CDNA5 ISA:
//   A 16x32: lanes 0-15 K{0-7,16-23}, lanes 16-31 K{8-15,24-31}
//   B 32x16: lane = column, element i -> K = 8*half + i (i<8), 8 + 8*half + i (i>=8)
//   C/D    : VGPR j -> row j + 8*half, column = lane&15
// ---------------------------------------------------------------------------
__global__ __launch_bounds__(256) void k_gemm_pw(
    const u16* __restrict__ A, const u16* __restrict__ W,
    u16* O, const u16* Res,
    const float* __restrict__ bng, const float* __restrict__ bnb,
    const float* __restrict__ bnm, const float* __restrict__ bnv,
    const float* __restrict__ bias,
    int NB, int Hin, int Win, int Hout, int Wout,
    int Cinp, int Coutp, int Cout, int stride, int relu)
{
    __shared__ u16 lsB[2][64 * BSTR];

    const int lane = threadIdx.x & 31;
    const int wave = threadIdx.x >> 5;
    const int half = lane >> 4;
    const int l16  = lane & 15;
    const long HW = (long)Hout * Wout;
    const long M  = (long)NB * HW;
    // No early return: all waves stage/barrier; OOB waves compute on clamped
    // rows and skip their stores.
    const long m0 = ((long)blockIdx.x * 8 + wave) * 32;
    const int n0 = blockIdx.y * 64;

    // Row pointers for the two 16-row A fragments of this lane (clamped).
    long r0 = m0 + l16;      if (r0 >= M) r0 = M - 1;
    long r1 = m0 + 16 + l16; if (r1 >= M) r1 = M - 1;
    const u16 *a0, *a1;
    if (stride == 1) {                    // Hout==Hin, Wout==Win: identity map
        a0 = A + r0 * Cinp;
        a1 = A + r1 * Cinp;
    } else {                              // stride-2 1x1 shortcut
        int ni = (int)(r0 / HW), re = (int)(r0 - (long)ni * HW);
        int ho = re / Wout, wo = re - ho * Wout;
        a0 = A + ((long)(ni * Hin + ho * 2) * Win + wo * 2) * Cinp;
        ni = (int)(r1 / HW); re = (int)(r1 - (long)ni * HW);
        ho = re / Wout; wo = re - ho * Wout;
        a1 = A + ((long)(ni * Hin + ho * 2) * Win + wo * 2) * Cinp;
    }
    a0 += 8 * half;                       // fold fragment offset into pointer
    a1 += 8 * half;

    // Cooperative B stage: 256 threads move the 64x32 bf16 tile (4KB) with
    // one global_load_b128 + ds_store_b128 each. row = n-col, seg = k-slice.
    const int srow = threadIdx.x >> 2;
    const int sseg = threadIdx.x & 3;
    int snc = n0 + srow; if (snc >= Coutp) snc = Coutp - 1;
    const u16* sW = W + (long)snc * Cinp + sseg * 8;
    u16* sL = &lsB[0][0] + srow * BSTR + sseg * 8;
    auto stageB = [&](int buf, int k) {
        *(u32x4*)(sL + buf * 64 * BSTR) = *(const u32x4*)(sW + k);
    };

    v8f acc[8] = {};
    stageB(0, 0);
    __syncthreads();
    int j = 0;
    for (int k = 0; k < Cinp; k += 32, ++j) {
        const int buf = j & 1;
        if (k + 32 < Cinp) stageB(buf ^ 1, k + 32);   // stage next tile
        __builtin_prefetch(a0 + k + 128, 0, 1);       // global_prefetch_b8
        __builtin_prefetch(a1 + k + 128, 0, 1);
        FragBF fa0, fa1;
        fa0.q[0] = *(const u32x4*)(a0 + k);
        fa0.q[1] = *(const u32x4*)(a0 + k + 16);
        fa1.q[0] = *(const u32x4*)(a1 + k);
        fa1.q[1] = *(const u32x4*)(a1 + k + 16);
        // Preload all four B fragments (8 ds_load_b128 in one clause, one
        // dscnt wait), then fire 8 back-to-back WMMAs.
        const u16* lb = &lsB[buf][8 * half];
        FragBF fb[4];
#pragma unroll
        for (int s = 0; s < 4; ++s) {
            const u16* lrow = lb + (s * 16 + l16) * BSTR;
            fb[s].q[0] = *(const u32x4*)(lrow);
            fb[s].q[1] = *(const u32x4*)(lrow + 16);
        }
#pragma unroll
        for (int s = 0; s < 4; ++s) {
            acc[s] = __builtin_amdgcn_wmma_f32_16x16x32_bf16(
                false, fa0.v, false, fb[s].v, (short)0, acc[s], false, false);
            acc[4 + s] = __builtin_amdgcn_wmma_f32_16x16x32_bf16(
                false, fa1.v, false, fb[s].v, (short)0, acc[4 + s], false, false);
        }
        __syncthreads();   // reads of buf done; stage of buf may proceed next
    }

#pragma unroll
    for (int s = 0; s < 4; ++s) {
        const int ncol = n0 + s * 16 + l16;
        if (ncol >= Coutp) continue;
        float scale = 0.f, shift = 0.f;
        if (ncol < Cout) {
            const float sc = bng[ncol] * rsqrtf(bnv[ncol] + 1e-5f);
            scale = sc;
            shift = bnb[ncol] - bnm[ncol] * sc;
            if (bias) shift += bias[ncol] * sc;   // bn(conv + bias)
        }
#pragma unroll
        for (int g = 0; g < 2; ++g) {
            const v8f av = acc[g * 4 + s];
#pragma unroll
            for (int jj = 0; jj < 8; ++jj) {
                const long mr = m0 + g * 16 + half * 8 + jj;
                if (mr >= M) continue;
                float vv = av[jj] * scale + shift;
                if (Res) vv += bf2f(Res[mr * (long)Coutp + ncol]);
                if (relu) vv = fmaxf(vv, 0.f);
                O[mr * (long)Coutp + ncol] = f2bf(vv);
            }
        }
    }
}

// ---------------------------------------------------------------------------
// 3x3 conv (stride 1, pad 1) as 9-tap accumulated GEMM (conv2 only).
// 16x64 wave tile, direct global weight loads.
// ---------------------------------------------------------------------------
__global__ __launch_bounds__(256) void k_conv3x3(
    const u16* __restrict__ A, const u16* __restrict__ W,
    u16* O,
    const float* __restrict__ bng, const float* __restrict__ bnb,
    const float* __restrict__ bnm, const float* __restrict__ bnv,
    int NB, int H, int Wd, int Cinp, int Coutp, int Cout, int relu)
{
    const int lane = threadIdx.x & 31;
    const int wave = threadIdx.x >> 5;
    const int half = lane >> 4;
    const int l16  = lane & 15;
    const long HW = (long)H * Wd;
    const long M  = (long)NB * HW;
    const long m0 = ((long)blockIdx.x * 8 + wave) * 16;
    if (m0 >= M) return;
    const int n0 = blockIdx.y * 64;

    long mrow = m0 + l16;
    if (mrow >= M) mrow = M - 1;
    const int nimg = (int)(mrow / HW);
    const int rem  = (int)(mrow - (long)nimg * HW);
    const int ho = rem / Wd;
    const int wo = rem - ho * Wd;

    const u16* wrow[4];
#pragma unroll
    for (int s = 0; s < 4; ++s) {
        int nc = n0 + s * 16 + l16;
        if (nc >= Coutp) nc = Coutp - 1;
        wrow[s] = W + (long)nc * Cinp + 8 * half;
    }
    const long tapstride = (long)Coutp * Cinp;

    v8f acc[4] = {};
    const u32x4 Z4 = {0u, 0u, 0u, 0u};
    int t = 0;
    for (int ty = 0; ty < 3; ++ty) {
        const int hi = ho + ty - 1;
        const bool vy = (hi >= 0) && (hi < H);
        for (int tx = 0; tx < 3; ++tx, ++t) {
            const int wi = wo + tx - 1;
            const bool valid = vy && (wi >= 0) && (wi < Wd);
            const u16* arow = A + ((long)(nimg * H + hi) * Wd + wi) * Cinp + 8 * half;
            for (int k = 0; k < Cinp; k += 32) {
                FragBF fa;
                if (valid) {
                    fa.q[0] = *(const u32x4*)(arow + k);
                    fa.q[1] = *(const u32x4*)(arow + k + 16);
                } else {
                    fa.q[0] = Z4; fa.q[1] = Z4;
                }
#pragma unroll
                for (int s = 0; s < 4; ++s) {
                    const u16* wr = wrow[s] + (long)t * tapstride;
                    FragBF fb;
                    fb.q[0] = *(const u32x4*)(wr + k);
                    fb.q[1] = *(const u32x4*)(wr + k + 16);
                    acc[s] = __builtin_amdgcn_wmma_f32_16x16x32_bf16(
                        false, fa.v, false, fb.v, (short)0, acc[s], false, false);
                }
            }
        }
    }

#pragma unroll
    for (int s = 0; s < 4; ++s) {
        const int ncol = n0 + s * 16 + l16;
        if (ncol >= Coutp) continue;
        float scale = 0.f, shift = 0.f;
        if (ncol < Cout) {
            const float sc = bng[ncol] * rsqrtf(bnv[ncol] + 1e-5f);
            scale = sc;
            shift = bnb[ncol] - bnm[ncol] * sc;
        }
#pragma unroll
        for (int jj = 0; jj < 8; ++jj) {
            const long mr = m0 + half * 8 + jj;
            if (mr >= M) continue;
            float vv = acc[s][jj] * scale + shift;
            if (relu) vv = fmaxf(vv, 0.f);
            O[mr * (long)Coutp + ncol] = f2bf(vv);
        }
    }
}

// ---------------------------------------------------------------------------
// Depthwise 3x3, pad 1, stride 1, optional input-ReLU. One pixel per block;
// pixel decode is block-uniform (scalar), channels contiguous across lanes.
// ---------------------------------------------------------------------------
__global__ void k_dwconv(const u16* __restrict__ in, const float* __restrict__ w,
                         u16* __restrict__ out,
                         int H, int Wd, int Cp, int C, int relu_in)
{
    const long pix = blockIdx.x;                 // n*H*W + y*W + x
    const int xw = (int)(pix % Wd);
    const long t2 = pix / Wd;
    const int y = (int)(t2 % H);
    const int n = (int)(t2 / H);
    const long obase = pix * Cp;
    for (int c = threadIdx.x; c < Cp; c += blockDim.x) {
        if (c >= C) { out[obase + c] = 0; continue; }
        float acc = 0.f;
        const float* wc = w + (size_t)c * 9;
#pragma unroll
        for (int ky = 0; ky < 3; ++ky) {
            const int yy = y + ky - 1;
            if (yy < 0 || yy >= H) continue;
#pragma unroll
            for (int kx = 0; kx < 3; ++kx) {
                const int xx = xw + kx - 1;
                if (xx < 0 || xx >= Wd) continue;
                float vv = bf2f(in[((long)(n * H + yy) * Wd + xx) * Cp + c]);
                if (relu_in) vv = fmaxf(vv, 0.f);
                acc += vv * wc[ky * 3 + kx];
            }
        }
        out[obase + c] = f2bf(acc);
    }
}

// MaxPool2d(3, stride=2, padding=1); one output pixel per block.
__global__ void k_maxpool(const u16* __restrict__ in, u16* __restrict__ out,
                          int Hin, int Win, int Hout, int Wout, int Cp)
{
    const long pix = blockIdx.x;                 // n*Hout*Wout + ho*Wout + wo
    const int wo = (int)(pix % Wout);
    const long t2 = pix / Wout;
    const int ho = (int)(t2 % Hout);
    const int n = (int)(t2 / Hout);
    const long obase = pix * Cp;
    for (int c = threadIdx.x; c < Cp; c += blockDim.x) {
        float m = -INFINITY;
        for (int ky = 0; ky < 3; ++ky) {
            const int hi = ho * 2 + ky - 1;
            if (hi < 0 || hi >= Hin) continue;
            for (int kx = 0; kx < 3; ++kx) {
                const int wi = wo * 2 + kx - 1;
                if (wi < 0 || wi >= Win) continue;
                m = fmaxf(m, bf2f(in[((long)(n * Hin + hi) * Win + wi) * Cp + c]));
            }
        }
        out[obase + c] = f2bf(m);
    }
}

// conv1: 3x3 stride 2, Cin=2 (K=18) + BN + ReLU. fp32 NCHW -> bf16 NHWC.
// One pixel per 32-thread block, thread = output channel.
__global__ void k_conv1(const float* __restrict__ x, const float* __restrict__ w,
                        const float* __restrict__ g, const float* __restrict__ b,
                        const float* __restrict__ m, const float* __restrict__ v,
                        u16* __restrict__ out)
{
    const int Ho = 255, Wo = 127, Hin = 512, Win = 256, Ci = 2;
    const long pix = blockIdx.x;                 // n*Ho*Wo + ho*Wo + wo
    const int wo = (int)(pix % Wo);
    const long t2 = pix / Wo;
    const int ho = (int)(t2 % Ho);
    const int n = (int)(t2 / Ho);
    const int co = threadIdx.x;                  // 32 channels
    float acc = 0.f;
#pragma unroll
    for (int ci = 0; ci < Ci; ++ci)
#pragma unroll
        for (int ky = 0; ky < 3; ++ky)
#pragma unroll
            for (int kx = 0; kx < 3; ++kx)
                acc += x[((size_t)(n * Ci + ci) * Hin + (ho * 2 + ky)) * Win + (wo * 2 + kx)]
                     * w[((co * Ci + ci) * 3 + ky) * 3 + kx];
    const float sc = g[co] * rsqrtf(v[co] + 1e-5f);
    out[pix * 32 + co] = f2bf(fmaxf(acc * sc + (b[co] - m[co] * sc), 0.f));
}

// fp32 (Cout,Cin[,3,3]) -> bf16 [taps][Coutp][Cinp], zero padded.
__global__ void k_wconv(const float* __restrict__ src, u16* __restrict__ dst,
                        int Cout, int Cin, int Cop, int Cip, int taps)
{
    size_t idx = (size_t)blockIdx.x * 256 + threadIdx.x;
    const size_t total = (size_t)taps * Cop * Cip;
    if (idx >= total) return;
    const int i = (int)(idx % Cip);
    size_t r = idx / Cip;
    const int o = (int)(r % Cop);
    const int t = (int)(r / Cop);
    float v = 0.f;
    if (o < Cout && i < Cin)
        v = (taps == 9) ? src[(size_t)(o * Cin + i) * 9 + t]
                        : src[(size_t)o * Cin + i];
    dst[idx] = f2bf(v);
}

// Global average pool: bf16 NHWC feat -> fp32 pooled (NB, C).
__global__ void k_avgpool(const u16* __restrict__ feat, float* __restrict__ pooled,
                          int HW, int C)
{
    const int n = blockIdx.x;
    for (int c = threadIdx.x; c < C; c += blockDim.x) {
        float s = 0.f;
        for (int p = 0; p < HW; ++p) s += bf2f(feat[((size_t)n * HW + p) * C + c]);
        pooled[n * C + c] = s / (float)HW;
    }
}

// Per-image head: logits, argmax, CAM, min/max normalize, threshold 0.3.
__global__ __launch_bounds__(256) void k_head(
    const float* __restrict__ pooled, const float* __restrict__ fcw,
    const float* __restrict__ fcb, const u16* __restrict__ feat,
    float* out, int HW, int C, int cam_off)
{
    const int n = blockIdx.x;
    const int t = threadIdx.x;
    __shared__ float red[256];
    __shared__ float cams[128];
    __shared__ float sh_l0, sh_l1, sh_mn, sh_mx;
    __shared__ int sh_pred;

    float p0 = 0.f, p1 = 0.f;
    for (int c = t; c < C; c += 256) {
        const float pv = pooled[n * C + c];
        p0 += pv * fcw[c];
        p1 += pv * fcw[C + c];
    }
    red[t] = p0; __syncthreads();
    for (int s = 128; s > 0; s >>= 1) { if (t < s) red[t] += red[t + s]; __syncthreads(); }
    if (t == 0) sh_l0 = red[0] + fcb[0];
    __syncthreads();
    red[t] = p1; __syncthreads();
    for (int s = 128; s > 0; s >>= 1) { if (t < s) red[t] += red[t + s]; __syncthreads(); }
    if (t == 0) {
        sh_l1 = red[0] + fcb[1];
        out[n * 2 + 0] = sh_l0;
        out[n * 2 + 1] = sh_l1;
        sh_pred = (sh_l1 > sh_l0) ? 1 : 0;   // first-max tie-break like jnp.argmax
    }
    __syncthreads();
    const float* wsel = fcw + sh_pred * C;
    if (t < HW) {
        float s = 0.f;
        const u16* fr = feat + (size_t)(n * HW + t) * C;
        for (int c = 0; c < C; ++c) s += wsel[c] * bf2f(fr[c]);
        cams[t] = s;
    }
    __syncthreads();
    if (t == 0) {
        float mn = cams[0], mx = cams[0];
        for (int i = 1; i < HW; ++i) { mn = fminf(mn, cams[i]); mx = fmaxf(mx, cams[i]); }
        sh_mn = mn; sh_mx = mx;
    }
    __syncthreads();
    if (t < HW) {
        const float norm = (cams[t] - sh_mn) / (sh_mx - sh_mn);
        out[cam_off + n * HW + t] = (norm < 0.3f) ? 0.f : 1.f;
    }
}

// ---------------------------------------------------------------------------
// Host orchestration
// ---------------------------------------------------------------------------
extern "C" void kernel_launch(void* const* d_in, const int* in_sizes, int n_in,
                              void* d_out, int out_size, void* d_ws, size_t ws_size,
                              hipStream_t stream)
{
    (void)n_in; (void)out_size; (void)ws_size;
    const float* x_in = (const float*)d_in[0];
    const int NB = in_sizes[0] / (2 * 512 * 256);   // 16

    // ---- parameter leaves, insertion order of setup_inputs() ----
    int ii = 2;   // skip 'x' (0) and scalar 'b' (1)
    auto F = [&]() { return (const float*)d_in[ii++]; };
    struct BN   { const float *g, *b, *m, *v; };
    struct Sep  { const float *dw, *pw; };
    struct Ent  { Sep s1; BN b1; Sep s2; BN b2; };
    struct Sh   { const float *w, *bias; BN bn; };
    struct Mid  { Sep s1; BN b1; Sep s2; BN b2; Sep s3; BN b3; };
    auto loadBN  = [&]() { BN r; r.g = F(); r.b = F(); r.m = F(); r.v = F(); return r; };
    auto loadSep = [&]() { Sep s; s.dw = F(); s.pw = F(); return s; };
    auto loadEnt = [&]() { Ent e; e.s1 = loadSep(); e.b1 = loadBN(); e.s2 = loadSep(); e.b2 = loadBN(); return e; };
    auto loadSh  = [&]() { Sh s; s.w = F(); s.bias = F(); s.bn = loadBN(); return s; };
    auto loadMid = [&]() { Mid m; m.s1 = loadSep(); m.b1 = loadBN(); m.s2 = loadSep();
                           m.b2 = loadBN(); m.s3 = loadSep(); m.b3 = loadBN(); return m; };

    const float* conv1_w = F(); BN bn1 = loadBN();
    const float* conv2_w = F(); BN bn2 = loadBN();
    Ent c3r = loadEnt(); Sh c3s = loadSh();
    Ent c4r = loadEnt(); Sh c4s = loadSh();
    Ent c5r = loadEnt(); Sh c5s = loadSh();
    Mid mid[8]; for (int i = 0; i < 8; ++i) mid[i] = loadMid();
    Ent eres = loadEnt();
    Sh  eshort = loadSh();
    Ent econv = loadEnt();
    const float* fc_w = F();
    const float* fc_b = F();

    // ---- workspace carve ----
    char* ws = (char*)d_ws;
    u16* warena = (u16*)ws;                               // 48 MB bf16 weights
    const size_t G = 136ull << 20;
    u16* g0 = (u16*)(ws + (48ull << 20));
    u16* g1 = (u16*)(ws + (48ull << 20) + G);
    u16* g2 = (u16*)(ws + (48ull << 20) + 2 * G);
    u16* g3 = (u16*)(ws + (48ull << 20) + 3 * G);         // 40 MB slab
    float* pooled = (float*)(ws + (48ull << 20) + 3 * G + (40ull << 20));

    auto pad32 = [](int c) { return (c + 31) / 32 * 32; };

    size_t wcur = 0;
    auto convw = [&](const float* src, int Cout, int Cin, int taps) -> u16* {
        const int Cop = pad32(Cout), Cip = pad32(Cin);
        u16* dst = warena + wcur;
        const size_t total = (size_t)taps * Cop * Cip;
        wcur += total;
        k_wconv<<<dim3((unsigned)((total + 255) / 256)), 256, 0, stream>>>(
            src, dst, Cout, Cin, Cop, Cip, taps);
        return dst;
    };

    // 1x1 GEMM (+ optional stride-2, bias, residual, relu)
    auto gemm = [&](const u16* A, const u16* W, u16* O, const u16* Res, BN bn,
                    const float* bias, int Hin, int Win, int Hout, int Wout,
                    int Cin, int Cout, int stride, int relu) {
        const int Cip = pad32(Cin), Cop = pad32(Cout);
        const long M = (long)NB * Hout * Wout;
        dim3 grid((unsigned)((M + 255) / 256), (unsigned)((Cop + 63) / 64));
        k_gemm_pw<<<grid, 256, 0, stream>>>(A, W, O, Res, bn.g, bn.b, bn.m, bn.v,
            bias, NB, Hin, Win, Hout, Wout, Cip, Cop, Cout, stride, relu);
    };
    auto dwc = [&](const u16* in, const float* w, u16* out, int H, int Wd, int C, int relu_in) {
        const int Cp = pad32(C);
        const int bs = Cp < 256 ? Cp : 256;
        k_dwconv<<<dim3((unsigned)((long)NB * H * Wd)), bs, 0, stream>>>(
            in, w, out, H, Wd, Cp, C, relu_in);
    };
    auto mpool = [&](const u16* in, u16* out, int Hin, int Win, int C) {
        const int Cp = pad32(C), Ho = (Hin - 1) / 2 + 1, Wo = (Win - 1) / 2 + 1;
        const int bs = Cp < 256 ? Cp : 256;
        k_maxpool<<<dim3((unsigned)((long)NB * Ho * Wo)), bs, 0, stream>>>(
            in, out, Hin, Win, Ho, Wo, Cp);
    };

    // ---- convert all GEMM weights to bf16 (padded) ----
    u16* w_conv2 = convw(conv2_w, 64, 32, 9);
    u16* w_c3p1 = convw(c3r.s1.pw, 128, 64, 1);
    u16* w_c3p2 = convw(c3r.s2.pw, 128, 128, 1);
    u16* w_c3sw = convw(c3s.w, 128, 64, 1);
    u16* w_c4p1 = convw(c4r.s1.pw, 256, 128, 1);
    u16* w_c4p2 = convw(c4r.s2.pw, 256, 256, 1);
    u16* w_c4sw = convw(c4s.w, 256, 128, 1);
    u16* w_c5p1 = convw(c5r.s1.pw, 728, 256, 1);
    u16* w_c5p2 = convw(c5r.s2.pw, 728, 728, 1);
    u16* w_c5sw = convw(c5s.w, 728, 256, 1);
    u16* w_mid[8][3];
    for (int i = 0; i < 8; ++i) {
        w_mid[i][0] = convw(mid[i].s1.pw, 728, 728, 1);
        w_mid[i][1] = convw(mid[i].s2.pw, 728, 728, 1);
        w_mid[i][2] = convw(mid[i].s3.pw, 728, 728, 1);
    }
    u16* w_e1 = convw(eres.s1.pw, 728, 728, 1);
    u16* w_e2 = convw(eres.s2.pw, 1024, 728, 1);
    u16* w_es = convw(eshort.w, 1024, 728, 1);
    u16* w_x1 = convw(econv.s1.pw, 1536, 1024, 1);
    u16* w_x2 = convw(econv.s2.pw, 512, 1536, 1);

    // ---- stem ----
    k_conv1<<<dim3((unsigned)((long)NB * 255 * 127)), 32, 0, stream>>>(
        x_in, conv1_w, bn1.g, bn1.b, bn1.m, bn1.v, g2);
    {   // conv2: 3x3 pad1 + BN + ReLU via 9-tap WMMA GEMM
        const long M = (long)NB * 255 * 127;
        dim3 grid((unsigned)((M + 127) / 128), 1);
        k_conv3x3<<<grid, 256, 0, stream>>>(g2, w_conv2, g0,
            bn2.g, bn2.b, bn2.m, bn2.v, NB, 255, 127, 32, 64, 64, 1);
    }

    // ---- entry block c3 (no pre-relu), x in g0 (64ch @255x127) ----
    dwc(g0, c3r.s1.dw, g1, 255, 127, 64, 0);
    gemm(g1, w_c3p1, g2, nullptr, c3r.b1, nullptr, 255, 127, 255, 127, 64, 128, 1, 1);
    dwc(g2, c3r.s2.dw, g1, 255, 127, 128, 0);
    gemm(g1, w_c3p2, g2, nullptr, c3r.b2, nullptr, 255, 127, 255, 127, 128, 128, 1, 0);
    mpool(g2, g3, 255, 127, 128);
    gemm(g0, w_c3sw, g1, g3, c3s.bn, c3s.bias, 255, 127, 128, 64, 64, 128, 2, 0);

    // ---- entry block c4 (pre-relu), x in g1 (128ch @128x64) ----
    dwc(g1, c4r.s1.dw, g0, 128, 64, 128, 1);
    gemm(g0, w_c4p1, g2, nullptr, c4r.b1, nullptr, 128, 64, 128, 64, 128, 256, 1, 1);
    dwc(g2, c4r.s2.dw, g0, 128, 64, 256, 0);
    gemm(g0, w_c4p2, g2, nullptr, c4r.b2, nullptr, 128, 64, 128, 64, 256, 256, 1, 0);
    mpool(g2, g3, 128, 64, 256);
    gemm(g1, w_c4sw, g0, g3, c4s.bn, c4s.bias, 128, 64, 64, 32, 128, 256, 2, 0);

    // ---- entry block c5 (pre-relu), x in g0 (256ch @64x32) ----
    dwc(g0, c5r.s1.dw, g1, 64, 32, 256, 1);
    gemm(g1, w_c5p1, g2, nullptr, c5r.b1, nullptr, 64, 32, 64, 32, 256, 728, 1, 1);
    dwc(g2, c5r.s2.dw, g1, 64, 32, 728, 0);
    gemm(g1, w_c5p2, g2, nullptr, c5r.b2, nullptr, 64, 32, 64, 32, 728, 728, 1, 0);
    mpool(g2, g3, 64, 32, 728);
    gemm(g0, w_c5sw, g1, g3, c5s.bn, c5s.bias, 64, 32, 32, 16, 256, 728, 2, 0);

    // ---- middle flow: 8 residual blocks, x stays in g1 (728ch @32x16) ----
    for (int i = 0; i < 8; ++i) {
        dwc(g1, mid[i].s1.dw, g0, 32, 16, 728, 1);
        gemm(g0, w_mid[i][0], g2, nullptr, mid[i].b1, nullptr, 32, 16, 32, 16, 728, 728, 1, 0);
        dwc(g2, mid[i].s2.dw, g0, 32, 16, 728, 1);
        gemm(g0, w_mid[i][1], g2, nullptr, mid[i].b2, nullptr, 32, 16, 32, 16, 728, 728, 1, 0);
        dwc(g2, mid[i].s3.dw, g0, 32, 16, 728, 1);
        // residual add fused; O aliases Res (each element read once then written)
        gemm(g0, w_mid[i][2], g1, g1, mid[i].b3, nullptr, 32, 16, 32, 16, 728, 728, 1, 0);
    }

    // ---- exit residual block, x in g1 ----
    dwc(g1, eres.s1.dw, g0, 32, 16, 728, 1);
    gemm(g0, w_e1, g2, nullptr, eres.b1, nullptr, 32, 16, 32, 16, 728, 728, 1, 1);
    dwc(g2, eres.s2.dw, g0, 32, 16, 728, 0);
    gemm(g0, w_e2, g2, nullptr, eres.b2, nullptr, 32, 16, 32, 16, 728, 1024, 1, 0);
    mpool(g2, g3, 32, 16, 1024);
    gemm(g1, w_es, g0, g3, eshort.bn, eshort.bias, 32, 16, 16, 8, 728, 1024, 2, 0);

    // ---- exit conv, x in g0 (1024ch @16x8) ----
    dwc(g0, econv.s1.dw, g1, 16, 8, 1024, 0);
    gemm(g1, w_x1, g2, nullptr, econv.b1, nullptr, 16, 8, 16, 8, 1024, 1536, 1, 1);
    dwc(g2, econv.s2.dw, g1, 16, 8, 1536, 0);
    gemm(g1, w_x2, g3, nullptr, econv.b2, nullptr, 16, 8, 16, 8, 1536, 512, 1, 1);  // feat -> g3

    // ---- head: avgpool, fc, argmax, CAM ----
    k_avgpool<<<dim3(NB), 256, 0, stream>>>(g3, pooled, 128, 512);
    k_head<<<dim3(NB), 256, 0, stream>>>(pooled, fc_w, fc_b, g3, (float*)d_out,
                                         128, 512, NB * 2);
}
